// EnergyGNNBase_77592879169749
// MI455X (gfx1250) — matrix operand order repaired and checked by reading, not compiled, not run on hardware
//
#include <hip/hip_runtime.h>
#include <math.h>

// ============================================================================
// EnergyGNN forward for MI455X (gfx1250, wave32, WMMA).
//
// Dominant cost: tall-skinny GEMMs (M up to 200k, K,N in {64,128}). These run
// through v_wmma_f32_16x16x32_bf16 with fp32 accumulate. Weights are converted
// fp32->bf16 while being staged (fragment-major) into LDS; activations are
// converted to bf16 in registers while A-fragments are built from 4
// unconditional global_load_b128 per k-tile (rows past M clamp their pointer:
// garbage there only feeds C rows that are never stored). Each GEMM therefore
// reads each activation element exactly once from HBM (~32 FLOP/B vs
// 23.3 TB/s -> memory bound, WMMA keeps compute off the critical path).
//
// Input layout assumption (JAX pytree flatten order, dict keys sorted):
//  d_in[0..3]  x_building(200000x17) x_cable_group(20000x4)
//              x_transformer(2000x3) x_adjacency_cluster(5000x4)   [f32]
//  d_in[4..6]  edge_b2cg(2x200000) edge_cg2t(2x20000) edge_adj(2x1000000) [i32]
//  d_in[7..9]  season, is_weekend, hour (1 elem each)              [i32]
//  d_in[10..98] params, sorted-key order:
//    10-13  enc_ac.{W,b,beta,g}      14-17 enc_b.{W,b,beta,g}
//    18-21  enc_cg.{...}             22-25 enc_t.{...}
//    26-27  ln_b.{beta,g}  28-29 ln_cg  30-31 ln_t
//    32+18*l .. : mp[l]: g1.{Wd,Ws,ad,as,b} g2.{Wd,Ws,ad,as,b}
//                        gcn.{W,b} s1.{Wl,Wr,b} s2.{Wl,Wr,b}
//    86-87 out_ac.{W,b}  88-89 out_b  90-91 out_cg  92-93 out_t
//    94-98 tmp.{W,b,day,hour,season}
// Output: concat(h_b 200000x128, h_cg 20000x128, h_t 2000x128, h_ac 5000x64)
// ============================================================================

#define NB_   200000
#define NCG_  20000
#define NT_   2000
#define NAC_  5000
#define EB2CG_ 200000
#define ECG2T_ 20000
#define EADJ_  1000000

typedef __attribute__((ext_vector_type(16))) __bf16 v16bf;
typedef __attribute__((ext_vector_type(8)))  float  v8f;

__device__ __forceinline__ float wave_sum32(float v) {
  #pragma unroll
  for (int m = 16; m > 0; m >>= 1) v += __shfl_xor(v, m, 32);
  return v;
}

__device__ __forceinline__ unsigned f2ord(float f) {
  unsigned u = __float_as_uint(f);
  return (u & 0x80000000u) ? ~u : (u | 0x80000000u);
}
__device__ __forceinline__ float ord2f(unsigned u) {
  return (u & 0x80000000u) ? __uint_as_float(u & 0x7fffffffu)
                           : __uint_as_float(~u);
}

// ---------------------------------------------------------------------------
// WMMA GEMM: D[M,N] = A[M,K] @ W[K,N] (+bias[N]) (+D_prev if accum)
// CONTRACT: K must be a multiple of 32 (true for all call sites: 64/128).
// NT = N/16 (templated so accumulators stay in VGPRs). Block = 256 thr = 8
// waves; each wave computes 16 rows x N cols. W staged in LDS fragment-major.
// ---------------------------------------------------------------------------
template <int NT>
__global__ __launch_bounds__(256) void gemm_bf16_wmma(
    const float* __restrict__ A, const float* __restrict__ Wm,
    const float* __restrict__ bias, float* __restrict__ D,
    int M, int K, int accum)
{
  constexpr int N = NT * 16;
  __shared__ __align__(32) __bf16 ldsW[16384];   // up to K=128, N=128
  const int tid = threadIdx.x;
  const int KT  = K >> 5;

  // Stage W (K x N) into fragment-major LDS layout:
  // frag f=(k/32)*NT+(n/16); lane=((k%16)/8)*16+(n%16); elem=(k%8)+8*((k%32)/16)
  const int total = K * N;
  for (int idx = tid; idx < total; idx += 256) {
    int k = idx / N, n = idx - k * N;
    float v = Wm[k * N + n];
    int f  = (k >> 5) * NT + (n >> 4);
    int ln = ((k >> 3) & 1) * 16 + (n & 15);
    ldsW[(f << 9) + (ln << 4) + ((k & 7) | ((k & 16) >> 1))] = (__bf16)v;
  }
  __syncthreads();

  const int lane  = tid & 31;
  const int wave  = tid >> 5;
  const int mBase = blockIdx.x * 128 + wave * 16;
  const int r     = lane & 15;
  const int grp   = lane >> 4;
  const int gr    = mBase + r;
  // Clamp out-of-range rows instead of guarding: their products only reach
  // C rows that are never stored, so unconditional vector loads are safe.
  const int grow  = (gr < M) ? gr : (M - 1);
  const float4* __restrict__ arow4 =
      reinterpret_cast<const float4*>(A + (size_t)grow * K);

  v8f acc[NT];
  const v8f vzero = {0.f, 0.f, 0.f, 0.f, 0.f, 0.f, 0.f, 0.f};
  #pragma unroll
  for (int nt = 0; nt < NT; ++nt) acc[nt] = vzero;

  for (int kt = 0; kt < KT; ++kt) {
    // A fragment, 16-bit A 16x32 layout:
    // lane<16:  elems0-7 = K kb+0..7,  elems8-15 = K kb+16..23
    // lane>=16: elems0-7 = K kb+8..15, elems8-15 = K kb+24..31
    const int kq = (kt << 3) + (grp << 1);   // float4 index of kb + grp*8
    float4 a0 = arow4[kq];
    float4 a1 = arow4[kq + 1];
    float4 a2 = arow4[kq + 4];               // +16 floats
    float4 a3 = arow4[kq + 5];
    v16bf afr;
    afr[0]  = (__bf16)a0.x; afr[1]  = (__bf16)a0.y;
    afr[2]  = (__bf16)a0.z; afr[3]  = (__bf16)a0.w;
    afr[4]  = (__bf16)a1.x; afr[5]  = (__bf16)a1.y;
    afr[6]  = (__bf16)a1.z; afr[7]  = (__bf16)a1.w;
    afr[8]  = (__bf16)a2.x; afr[9]  = (__bf16)a2.y;
    afr[10] = (__bf16)a2.z; afr[11] = (__bf16)a2.w;
    afr[12] = (__bf16)a3.x; afr[13] = (__bf16)a3.y;
    afr[14] = (__bf16)a3.z; afr[15] = (__bf16)a3.w;
    #pragma unroll
    for (int nt = 0; nt < NT; ++nt) {
      v16bf bfr = *reinterpret_cast<const v16bf*>(
          &ldsW[((kt * NT + nt) << 9) + (lane << 4)]);
      acc[nt] = __builtin_amdgcn_wmma_f32_16x16x32_bf16(
          false, afr, false, bfr, (short)0, acc[nt], false, false);
    }
  }

  // C/D layout: VGPR j -> row mBase + grp*8 + j, col nt*16 + (lane&15)
  const int rb = mBase + grp * 8;
  #pragma unroll
  for (int nt = 0; nt < NT; ++nt) {
    const int col = nt * 16 + r;
    const float bv = bias ? bias[col] : 0.0f;
    #pragma unroll
    for (int j = 0; j < 8; ++j) {
      int rr = rb + j;
      if (rr < M) {
        size_t o = (size_t)rr * N + col;
        float v = acc[nt][j] + bv;
        if (accum) v += D[o];
        D[o] = v;
      }
    }
  }
}

// ---------------------------------------------------------------------------
// Encoder: Y = relu(layernorm(X@W + b)*g + beta).  K tiny (<=17), so scalar.
// One wave per row; NPL = N/32 channels per lane.
// ---------------------------------------------------------------------------
template <int NPL>
__global__ __launch_bounds__(256) void encode_kernel(
    const float* __restrict__ X, const float* __restrict__ Wm,
    const float* __restrict__ b, const float* __restrict__ g,
    const float* __restrict__ beta, float* __restrict__ Y, int M, int K)
{
  constexpr int N = NPL * 32;
  const int lane = threadIdx.x & 31;
  const int wave = threadIdx.x >> 5;
  const int row = blockIdx.x * 8 + wave;
  if (row >= M) return;            // uniform per wave
  const float* x = X + (size_t)row * K;
  float acc[NPL];
  #pragma unroll
  for (int p = 0; p < NPL; ++p) acc[p] = b[lane * NPL + p];
  for (int k = 0; k < K; ++k) {
    float xv = x[k];
    #pragma unroll
    for (int p = 0; p < NPL; ++p) acc[p] += xv * Wm[k * N + lane * NPL + p];
  }
  float s = 0.f;
  #pragma unroll
  for (int p = 0; p < NPL; ++p) s += acc[p];
  float mean = wave_sum32(s) / (float)N;
  float q = 0.f;
  #pragma unroll
  for (int p = 0; p < NPL; ++p) { float d = acc[p] - mean; q += d * d; }
  float rstd = rsqrtf(wave_sum32(q) / (float)N + 1e-5f);
  #pragma unroll
  for (int p = 0; p < NPL; ++p) {
    int c = lane * NPL + p;
    float v = (acc[p] - mean) * rstd * g[c] + beta[c];
    Y[(size_t)row * N + c] = fmaxf(v, 0.0f);
  }
}

// H = layernorm(H0 + H)*g + beta, N=128, one wave per row
__global__ __launch_bounds__(256) void ln_residual_kernel(
    float* __restrict__ H, const float* __restrict__ H0,
    const float* __restrict__ g, const float* __restrict__ beta, int M)
{
  const int lane = threadIdx.x & 31;
  const int wave = threadIdx.x >> 5;
  const int row = blockIdx.x * 8 + wave;
  if (row >= M) return;
  size_t base = (size_t)row * 128 + lane * 4;
  float v[4];
  float s = 0.f;
  #pragma unroll
  for (int i = 0; i < 4; ++i) { v[i] = H[base + i] + H0[base + i]; s += v[i]; }
  float mean = wave_sum32(s) / 128.0f;
  float q = 0.f;
  #pragma unroll
  for (int i = 0; i < 4; ++i) { float d = v[i] - mean; q += d * d; }
  float rstd = rsqrtf(wave_sum32(q) / 128.0f + 1e-5f);
  #pragma unroll
  for (int i = 0; i < 4; ++i) {
    int c = lane * 4 + i;
    H[base + i] = (v[i] - mean) * rstd * g[c] + beta[c];
  }
}

// a[n,h] = sum_c hs[n, h*32+c] * av[h*32+c]
__global__ void ash_kernel(const float* __restrict__ hs,
                           const float* __restrict__ av,
                           float* __restrict__ out, int n)
{
  int idx = blockIdx.x * blockDim.x + threadIdx.x;
  if (idx >= n * 4) return;
  int node = idx >> 2, h = idx & 3;
  const float* p = hs + (size_t)node * 128 + h * 32;
  const float* a = av + h * 32;
  float s = 0.f;
  #pragma unroll
  for (int c = 0; c < 32; ++c) s += p[c] * a[c];
  out[idx] = s;
}

__global__ void edge_e_kernel(const int* __restrict__ src,
                              const int* __restrict__ dst,
                              const float* __restrict__ as_,
                              const float* __restrict__ ad_,
                              float* __restrict__ e, unsigned* __restrict__ m,
                              int E)
{
  int idx = blockIdx.x * blockDim.x + threadIdx.x;
  if (idx >= E * 4) return;
  int i = idx >> 2, h = idx & 3;
  float v = as_[src[i] * 4 + h] + ad_[dst[i] * 4 + h];
  v = v > 0.f ? v : 0.2f * v;              // leaky_relu 0.2
  e[idx] = v;
  atomicMax(&m[dst[i] * 4 + h], f2ord(v));
}

__global__ void max_fin_kernel(unsigned* __restrict__ mu, int n) {
  int idx = blockIdx.x * blockDim.x + threadIdx.x;
  if (idx >= n) return;
  float f = ord2f(mu[idx]);
  if (!__builtin_isfinite(f)) f = 0.f;
  ((float*)mu)[idx] = f;
}

__global__ void edge_exp_kernel(const int* __restrict__ dst,
                                float* __restrict__ e,
                                const float* __restrict__ m,
                                float* __restrict__ den, int E)
{
  int idx = blockIdx.x * blockDim.x + threadIdx.x;
  if (idx >= E * 4) return;
  int i = idx >> 2, h = idx & 3;
  float ex = expf(e[idx] - m[dst[i] * 4 + h]);
  e[idx] = ex;
  atomicAdd(&den[dst[i] * 4 + h], ex);
}

__global__ void gat_scatter_kernel(const int* __restrict__ src,
                                   const int* __restrict__ dst,
                                   const float* __restrict__ ex,
                                   const float* __restrict__ den,
                                   const float* __restrict__ hs,
                                   float* __restrict__ out, int E)
{
  int idx = blockIdx.x * blockDim.x + threadIdx.x;
  if (idx >= E * 128) return;
  int i = idx >> 7, c = idx & 127, h = c >> 5;
  int s = src[i], d = dst[i];
  float alpha = ex[i * 4 + h] / (den[d * 4 + h] + 1e-16f);
  atomicAdd(&out[(size_t)d * 128 + c], hs[(size_t)s * 128 + c] * alpha);
}

// H[n,c] += scale * (T[n,c] + b[c])
__global__ void add_scaled_bias_kernel(float* __restrict__ H,
                                       const float* __restrict__ T,
                                       const float* __restrict__ b,
                                       float scale, int n)
{
  int idx = blockIdx.x * blockDim.x + threadIdx.x;
  if (idx >= n) return;
  H[idx] += scale * (T[idx] + b[idx & 127]);
}

__global__ void sage_scatter_kernel(const int* __restrict__ src,
                                    const int* __restrict__ dst,
                                    const float* __restrict__ xs,
                                    float* __restrict__ agg, int E)
{
  int idx = blockIdx.x * blockDim.x + threadIdx.x;
  if (idx >= E * 128) return;
  int i = idx >> 7, c = idx & 127;
  atomicAdd(&agg[(size_t)dst[i] * 128 + c], xs[(size_t)src[i] * 128 + c]);
}

__global__ void cnt_kernel(const int* __restrict__ dst,
                           float* __restrict__ cnt, int E)
{
  int idx = blockIdx.x * blockDim.x + threadIdx.x;
  if (idx >= E) return;
  atomicAdd(&cnt[dst[idx]], 1.0f);
}

__global__ void sage_div_kernel(float* __restrict__ agg,
                                const float* __restrict__ cnt, int n)
{
  int idx = blockIdx.x * blockDim.x + threadIdx.x;
  if (idx >= n) return;
  agg[idx] /= fmaxf(cnt[idx >> 7], 1.0f);
}

__global__ void deg_scatter_kernel(const int* __restrict__ col,
                                   float* __restrict__ deg, int E)
{
  int idx = blockIdx.x * blockDim.x + threadIdx.x;
  if (idx >= E) return;
  atomicAdd(&deg[col[idx]], 1.0f);
}

__global__ void dinv_kernel(float* __restrict__ deg, int n) {
  int idx = blockIdx.x * blockDim.x + threadIdx.x;
  if (idx >= n) return;
  float d = deg[idx];
  deg[idx] = (d > 0.f) ? rsqrtf(fmaxf(d, 1e-12f)) : 0.f;
}

__global__ void gcn_scatter_kernel(const int* __restrict__ row,
                                   const int* __restrict__ col,
                                   const float* __restrict__ dinv,
                                   const float* __restrict__ xw,
                                   float* __restrict__ out, int E)
{
  int idx = blockIdx.x * blockDim.x + threadIdx.x;
  if (idx >= E * 128) return;
  int i = idx >> 7, c = idx & 127;
  int rr = row[i], cc = col[i];
  float nm = dinv[rr] * dinv[cc];
  atomicAdd(&out[(size_t)cc * 128 + c], nm * xw[(size_t)rr * 128 + c]);
}

// self-loop term folded in: H += 0.2*(out + dinv^2*xw + b)
__global__ void gcn_epilogue_kernel(float* __restrict__ H,
                                    const float* __restrict__ out,
                                    const float* __restrict__ xw,
                                    const float* __restrict__ dinv,
                                    const float* __restrict__ b, int n)
{
  int idx = blockIdx.x * blockDim.x + threadIdx.x;
  if (idx >= n) return;
  int nn = idx >> 7, c = idx & 127;
  H[idx] += 0.2f * (out[idx] + dinv[nn] * dinv[nn] * xw[idx] + b[c]);
}

__global__ void fill_u32_kernel(unsigned* __restrict__ p, unsigned v, int n) {
  int idx = blockIdx.x * blockDim.x + threadIdx.x;
  if (idx < n) p[idx] = v;
}

// temporal embedding -> bias vectors for out_b / out_cg heads
__global__ void temporal_kernel(const int* __restrict__ season,
                                const int* __restrict__ wkd,
                                const int* __restrict__ hour,
                                const float* __restrict__ Wseason,
                                const float* __restrict__ Wday,
                                const float* __restrict__ Whour,
                                const float* __restrict__ W36,
                                const float* __restrict__ b36,
                                const float* __restrict__ Wb,
                                const float* __restrict__ bb,
                                const float* __restrict__ Wcg,
                                const float* __restrict__ bcg,
                                float* __restrict__ vecb,
                                float* __restrict__ veccg)
{
  __shared__ float te[36], te2[36];
  int t = threadIdx.x;
  if (t < 12)       te[t] = Wseason[season[0] * 12 + t];
  else if (t < 20)  te[t] = Wday[wkd[0] * 8 + (t - 12)];
  else if (t < 36)  te[t] = Whour[hour[0] * 16 + (t - 20)];
  __syncthreads();
  if (t < 36) {
    float a = b36[t];
    for (int j = 0; j < 36; ++j) a += te[j] * W36[j * 36 + t];
    te2[t] = fmaxf(a, 0.f);
  }
  __syncthreads();
  if (t < 128) {
    float a = bb[t], c = bcg[t];
    for (int j = 0; j < 36; ++j) {
      a += te2[j] * Wb[(128 + j) * 128 + t];
      c += te2[j] * Wcg[(128 + j) * 128 + t];
    }
    vecb[t] = a;
    veccg[t] = c;
  }
}

// ===========================================================================
extern "C" void kernel_launch(void* const* d_in, const int* in_sizes, int n_in,
                              void* d_out, int out_size, void* d_ws,
                              size_t ws_size, hipStream_t stream)
{
  (void)in_sizes; (void)n_in; (void)out_size; (void)ws_size;
  #define PF(i) ((const float*)d_in[(i)])

  const float* xb  = PF(0);
  const float* xcg = PF(1);
  const float* xt  = PF(2);
  const float* xac = PF(3);
  const int* eb = (const int*)d_in[4];   // row0: b src, row1: cg dst
  const int* ec = (const int*)d_in[5];   // row0: cg src, row1: t dst
  const int* ea = (const int*)d_in[6];   // row0, row1
  const int* season = (const int*)d_in[7];
  const int* wkd    = (const int*)d_in[8];
  const int* hour   = (const int*)d_in[9];

  // ---- workspace layout (floats) ----
  float* W0 = (float*)d_ws;
  size_t off = 0;
  auto AL = [&](size_t n) { float* p = W0 + off; off += n; return p; };
  float* h_b  = AL((size_t)NB_ * 128);
  float* h_cg = AL((size_t)NCG_ * 128);
  float* h_t  = AL((size_t)NT_ * 128);
  float* h_ac = AL((size_t)NAC_ * 64);
  float* hb0  = AL((size_t)NB_ * 128);
  float* hcg0 = AL((size_t)NCG_ * 128);
  float* ht0  = AL((size_t)NT_ * 128);
  float* tb1  = AL((size_t)NB_ * 128);
  float* tb2  = AL((size_t)NB_ * 128);
  float* tcg1 = AL((size_t)NCG_ * 128);
  float* tcg2 = AL((size_t)NCG_ * 128);
  float* tt1  = AL((size_t)NT_ * 128);
  float* tt2  = AL((size_t)NT_ * 128);
  float* asb  = AL((size_t)NB_ * 4);
  float* adcg = AL((size_t)NCG_ * 4);
  float* ascg = AL((size_t)NCG_ * 4);
  float* adt  = AL((size_t)NT_ * 4);
  float* e1   = AL((size_t)EB2CG_ * 4);
  float* mcg  = AL((size_t)NCG_ * 4);
  float* dencg= AL((size_t)NCG_ * 4);
  float* e2   = AL((size_t)ECG2T_ * 4);
  float* mt   = AL((size_t)NT_ * 4);
  float* dent = AL((size_t)NT_ * 4);
  float* cntcg= AL((size_t)NCG_);
  float* cntb = AL((size_t)NB_);
  float* degb = AL((size_t)NB_);
  float* vecb = AL(128);
  float* veccg= AL(128);

  float* ob  = (float*)d_out;
  float* ocg = ob + (size_t)NB_ * 128;
  float* ot  = ocg + (size_t)NCG_ * 128;
  float* oac = ot + (size_t)NT_ * 128;

  auto G1 = [](long long n) { return dim3((unsigned)((n + 255) / 256)); };
  auto gemm128 = [&](const float* A, const float* Wm, const float* bias,
                     float* D, int M, int K, int accum) {
    gemm_bf16_wmma<8><<<dim3((unsigned)((M + 127) / 128)), 256, 0, stream>>>(
        A, Wm, bias, D, M, K, accum);
  };
  auto gemm64 = [&](const float* A, const float* Wm, const float* bias,
                    float* D, int M, int K, int accum) {
    gemm_bf16_wmma<4><<<dim3((unsigned)((M + 127) / 128)), 256, 0, stream>>>(
        A, Wm, bias, D, M, K, accum);
  };

  // ---- encoders ----
  encode_kernel<4><<<G1((long long)NB_ * 32), 256, 0, stream>>>(
      xb, PF(14), PF(15), PF(17), PF(16), h_b, NB_, 17);
  encode_kernel<4><<<G1((long long)NCG_ * 32), 256, 0, stream>>>(
      xcg, PF(18), PF(19), PF(21), PF(20), h_cg, NCG_, 4);
  encode_kernel<4><<<G1((long long)NT_ * 32), 256, 0, stream>>>(
      xt, PF(22), PF(23), PF(25), PF(24), h_t, NT_, 3);
  encode_kernel<2><<<G1((long long)NAC_ * 32), 256, 0, stream>>>(
      xac, PF(10), PF(11), PF(13), PF(12), h_ac, NAC_, 4);

  // GAT: hs=xsrc@Ws, hd=xdst@Wd, edge softmax by dst, scatter, residual.
  auto run_gat = [&](const float* xsrc, float* hdst, const int* src,
                     const int* dst, int E, int Msrc, int Mdst,
                     const float* Ws, const float* Wd, const float* av_s,
                     const float* av_d, const float* bg, float* hsb,
                     float* hdb, float* asB, float* adB, float* eB, float* mB,
                     float* denB, float* outB) {
    gemm128(xsrc, Ws, nullptr, hsb, Msrc, 128, 0);
    gemm128(hdst, Wd, nullptr, hdb, Mdst, 128, 0);
    ash_kernel<<<G1((long long)Msrc * 4), 256, 0, stream>>>(hsb, av_s, asB, Msrc);
    ash_kernel<<<G1((long long)Mdst * 4), 256, 0, stream>>>(hdb, av_d, adB, Mdst);
    fill_u32_kernel<<<G1((long long)Mdst * 4), 256, 0, stream>>>(
        (unsigned*)mB, 0x007FFFFFu /* f2ord(-inf) */, Mdst * 4);
    hipMemsetAsync(denB, 0, (size_t)Mdst * 4 * sizeof(float), stream);
    hipMemsetAsync(outB, 0, (size_t)Mdst * 128 * sizeof(float), stream);
    edge_e_kernel<<<G1((long long)E * 4), 256, 0, stream>>>(
        src, dst, asB, adB, eB, (unsigned*)mB, E);
    max_fin_kernel<<<G1((long long)Mdst * 4), 256, 0, stream>>>(
        (unsigned*)mB, Mdst * 4);
    edge_exp_kernel<<<G1((long long)E * 4), 256, 0, stream>>>(dst, eB, mB, denB, E);
    gat_scatter_kernel<<<G1((long long)E * 128), 256, 0, stream>>>(
        src, dst, eB, denB, hsb, outB, E);
    add_scaled_bias_kernel<<<G1((long long)Mdst * 128), 256, 0, stream>>>(
        hdst, outB, bg, 0.5f, Mdst * 128);
  };

  for (int l = 0; l < 3; ++l) {
    const int L = 32 + 18 * l;
    // snapshots for residual layernorm
    hipMemcpyAsync(hb0, h_b, (size_t)NB_ * 128 * sizeof(float),
                   hipMemcpyDeviceToDevice, stream);
    hipMemcpyAsync(hcg0, h_cg, (size_t)NCG_ * 128 * sizeof(float),
                   hipMemcpyDeviceToDevice, stream);
    hipMemcpyAsync(ht0, h_t, (size_t)NT_ * 128 * sizeof(float),
                   hipMemcpyDeviceToDevice, stream);

    // g1: b -> cg   (Wd=L+0, Ws=L+1, ad=L+2, as=L+3, b=L+4)
    run_gat(h_b, h_cg, eb, eb + EB2CG_, EB2CG_, NB_, NCG_,
            PF(L + 1), PF(L + 0), PF(L + 3), PF(L + 2), PF(L + 4),
            tb1, tcg1, asb, adcg, e1, mcg, dencg, tcg2);
    // g2: cg -> t   (Wd=L+5, Ws=L+6, ad=L+7, as=L+8, b=L+9)
    run_gat(h_cg, h_t, ec, ec + ECG2T_, ECG2T_, NCG_, NT_,
            PF(L + 6), PF(L + 5), PF(L + 8), PF(L + 7), PF(L + 9),
            tcg1, tt1, ascg, adt, e2, mt, dent, tt2);

    // s1: t -> cg  (src = ec+E (t), dst = ec (cg))  Wl=L+12 Wr=L+13 b=L+14
    hipMemsetAsync(tcg1, 0, (size_t)NCG_ * 128 * sizeof(float), stream);
    hipMemsetAsync(cntcg, 0, (size_t)NCG_ * sizeof(float), stream);
    sage_scatter_kernel<<<G1((long long)ECG2T_ * 128), 256, 0, stream>>>(
        ec + ECG2T_, ec, h_t, tcg1, ECG2T_);
    cnt_kernel<<<G1(ECG2T_), 256, 0, stream>>>(ec, cntcg, ECG2T_);
    sage_div_kernel<<<G1((long long)NCG_ * 128), 256, 0, stream>>>(
        tcg1, cntcg, NCG_ * 128);
    gemm128(tcg1, PF(L + 12), nullptr, tcg2, NCG_, 128, 0);
    gemm128(h_cg, PF(L + 13), nullptr, tcg2, NCG_, 128, 1);
    add_scaled_bias_kernel<<<G1((long long)NCG_ * 128), 256, 0, stream>>>(
        h_cg, tcg2, PF(L + 14), 0.3f, NCG_ * 128);

    // s2: cg -> b  (src = eb+E (cg), dst = eb (b))  Wl=L+15 Wr=L+16 b=L+17
    hipMemsetAsync(tb1, 0, (size_t)NB_ * 128 * sizeof(float), stream);
    hipMemsetAsync(cntb, 0, (size_t)NB_ * sizeof(float), stream);
    sage_scatter_kernel<<<G1((long long)EB2CG_ * 128), 256, 0, stream>>>(
        eb + EB2CG_, eb, h_cg, tb1, EB2CG_);
    cnt_kernel<<<G1(EB2CG_), 256, 0, stream>>>(eb, cntb, EB2CG_);
    sage_div_kernel<<<G1((long long)NB_ * 128), 256, 0, stream>>>(
        tb1, cntb, NB_ * 128);
    gemm128(tb1, PF(L + 15), nullptr, tb2, NB_, 128, 0);
    gemm128(h_b, PF(L + 16), nullptr, tb2, NB_, 128, 1);
    add_scaled_bias_kernel<<<G1((long long)NB_ * 128), 256, 0, stream>>>(
        h_b, tb2, PF(L + 17), 0.3f, NB_ * 128);

    // gcn on b  (W=L+10, b=L+11); self loops folded in (deg init 1, epilogue)
    fill_u32_kernel<<<G1(NB_), 256, 0, stream>>>((unsigned*)degb, 0x3f800000u,
                                                 NB_);
    deg_scatter_kernel<<<G1(EADJ_), 256, 0, stream>>>(ea + EADJ_, degb, EADJ_);
    dinv_kernel<<<G1(NB_), 256, 0, stream>>>(degb, NB_);
    gemm128(h_b, PF(L + 10), nullptr, tb1, NB_, 128, 0);
    hipMemsetAsync(tb2, 0, (size_t)NB_ * 128 * sizeof(float), stream);
    gcn_scatter_kernel<<<G1((long long)EADJ_ * 128), 256, 0, stream>>>(
        ea, ea + EADJ_, degb, tb1, tb2, EADJ_);
    gcn_epilogue_kernel<<<G1((long long)NB_ * 128), 256, 0, stream>>>(
        h_b, tb2, tb1, degb, PF(L + 11), NB_ * 128);

    // residual layernorms (ln_b g=27 beta=26, ln_cg 29/28, ln_t 31/30)
    ln_residual_kernel<<<G1((long long)NB_ * 32), 256, 0, stream>>>(
        h_b, hb0, PF(27), PF(26), NB_);
    ln_residual_kernel<<<G1((long long)NCG_ * 32), 256, 0, stream>>>(
        h_cg, hcg0, PF(29), PF(28), NCG_);
    ln_residual_kernel<<<G1((long long)NT_ * 32), 256, 0, stream>>>(
        h_t, ht0, PF(31), PF(30), NT_);
  }

  // temporal embedding -> bias vecs (te@W[128:164] + b), absorbing the concat
  temporal_kernel<<<1, 128, 0, stream>>>(
      season, wkd, hour, PF(98), PF(96), PF(97), PF(94), PF(95),
      PF(88), PF(89), PF(90), PF(91), vecb, veccg);

  // output heads
  gemm128(h_b, PF(88), vecb, ob, NB_, 128, 0);    // uses W rows 0..127
  gemm128(h_cg, PF(90), veccg, ocg, NCG_, 128, 0);
  gemm128(h_t, PF(92), PF(93), ot, NT_, 128, 0);
  gemm64(h_ac, PF(86), PF(87), oac, NAC_, 64, 0);

  #undef PF
}
